// GraphConditionedPolicy_18940805775871
// MI455X (gfx1250) — compile-verified, matrix operand order checked by reading, not run on hardware
//
#include <hip/hip_runtime.h>
#include <hip/hip_bf16.h>
#include <stdint.h>

// Problem constants (match reference)
#define A_N    8
#define B_N    4096
#define OBS_N  512
#define ACT_N  64
#define H_N    1024
#define G_N    64
#define NN_N   10000
#define E_N    40000
#define NODE_N 32
#define GH_N   64
#define AEMB_N 16

typedef __attribute__((ext_vector_type(16))) __bf16 v16bf;
typedef __attribute__((ext_vector_type(8)))  __bf16 v8bf;
typedef __attribute__((ext_vector_type(8)))  float  v8f;

__device__ __forceinline__ unsigned short f2bf(float f) {
  unsigned u = __float_as_uint(f);
  u += 0x7FFFu + ((u >> 16) & 1u);           // round-to-nearest-even
  return (unsigned short)(u >> 16);
}
__device__ __forceinline__ float bf2f(unsigned short h) {
  return __uint_as_float(((unsigned)h) << 16);
}

// CDNA5 async global->LDS copy (ASYNCcnt-tracked). Generic pointers to
// __shared__ carry the LDS byte offset in their low 32 bits.
__device__ __forceinline__ void async_ld_b128(void* lds, const void* gaddr) {
  unsigned ldsOff = (unsigned)(unsigned long long)lds;
  asm volatile("global_load_async_to_lds_b128 %0, %1, off"
               :: "v"(ldsOff), "v"(gaddr) : "memory");
}
__device__ __forceinline__ void wait_async0() {
#if __has_builtin(__builtin_amdgcn_s_wait_asynccnt)
  __builtin_amdgcn_s_wait_asynccnt(0);
#else
  asm volatile("s_wait_asynccnt 0x0" ::: "memory");
#endif
}

// ---------------------------------------------------------------- utilities
__global__ void zero_f32(float* p, int n) {
  int i = blockIdx.x * blockDim.x + threadIdx.x;
  if (i < n) p[i] = 0.0f;
}

// f32 [K][Nn]  ->  bf16 [Nn][K]   (N-major weight for WMMA B fragments)
__global__ void transpose_to_bf16(const float* __restrict__ src, unsigned short* __restrict__ dst,
                                  int K, int Nn) {
  int idx = blockIdx.x * blockDim.x + threadIdx.x;
  if (idx >= K * Nn) return;
  int n = idx / K, k = idx - n * K;
  dst[idx] = f2bf(src[(long long)k * Nn + n]);
}

__global__ void f32_to_bf16_kernel(const float* __restrict__ src, unsigned short* __restrict__ dst, int n) {
  int i = blockIdx.x * blockDim.x + threadIdx.x;
  if (i < n) dst[i] = f2bf(src[i]);
}

// ---------------------------------------------------------------- GNN (tiny, VALU f32)
__global__ __launch_bounds__(64) void gnn_l0(const int* __restrict__ nt, const int* __restrict__ na,
                       const float* __restrict__ te, const float* __restrict__ ge,
                       const float* __restrict__ W, const float* __restrict__ b,
                       const float* __restrict__ g, const float* __restrict__ be,
                       float* __restrict__ h0) {
  __shared__ float x[64];
  __shared__ float pre[64];
  int n = blockIdx.x, t = threadIdx.x;
  if (t < 32) x[t] = te[nt[n] * 32 + t];
  else        x[t] = ge[na[n] * 32 + (t - 32)];
  __syncthreads();
  float s = b[t];
#pragma unroll 8
  for (int i = 0; i < 64; ++i) s += x[i] * W[i * 64 + t];
  s = fmaxf(s, 0.0f);
  pre[t] = s;
  __syncthreads();
  float m = 0.f;
  for (int i = 0; i < 64; ++i) m += pre[i];
  m *= (1.0f / 64.0f);
  float v = 0.f;
  for (int i = 0; i < 64; ++i) { float d = pre[i] - m; v += d * d; }
  v *= (1.0f / 64.0f);
  h0[n * 64 + t] = (s - m) * rsqrtf(v + 1e-5f) * g[t] + be[t];
}

__global__ void edge_scatter(const int* __restrict__ edges, const float* __restrict__ h0,
                             float* __restrict__ agg, float* __restrict__ deg) {
  int tid = blockIdx.x * blockDim.x + threadIdx.x;
  if (tid >= E_N * 64) return;
  int e = tid >> 6, c = tid & 63;
  int s = edges[2 * e], d = edges[2 * e + 1];
  atomicAdd(&agg[(long long)d * 64 + c], h0[(long long)s * 64 + c]);
  atomicAdd(&agg[(long long)s * 64 + c], h0[(long long)d * 64 + c]);
  if (c == 0) { atomicAdd(&deg[d], 1.0f); atomicAdd(&deg[s], 1.0f); }
}

__global__ __launch_bounds__(64) void gnn_l1(const float* __restrict__ h0, const float* __restrict__ agg,
                       const float* __restrict__ deg,
                       const float* __restrict__ W, const float* __restrict__ b,
                       const float* __restrict__ g, const float* __restrict__ be,
                       float* __restrict__ h1, float* __restrict__ colsum) {
  __shared__ float in[128];
  __shared__ float pre[64];
  int n = blockIdx.x, t = threadIdx.x;
  float dg = fmaxf(deg[n], 1.0f);
  in[t]      = h0[n * 64 + t];
  in[64 + t] = agg[n * 64 + t] / dg;
  __syncthreads();
  float s = b[t];
#pragma unroll 8
  for (int i = 0; i < 128; ++i) s += in[i] * W[i * 64 + t];
  s = fmaxf(s, 0.0f);
  pre[t] = s;
  __syncthreads();
  float m = 0.f;
  for (int i = 0; i < 64; ++i) m += pre[i];
  m *= (1.0f / 64.0f);
  float v = 0.f;
  for (int i = 0; i < 64; ++i) { float d = pre[i] - m; v += d * d; }
  v *= (1.0f / 64.0f);
  float o = (s - m) * rsqrtf(v + 1e-5f) * g[t] + be[t];
  h1[n * 64 + t] = o;
  atomicAdd(&colsum[t], o);
}

__global__ __launch_bounds__(64) void graph_proj(const float* __restrict__ colsum, const float* __restrict__ W,
                           const float* __restrict__ b, float* __restrict__ ge) {
  __shared__ float mean[64];
  int t = threadIdx.x;
  mean[t] = colsum[t] * (1.0f / (float)NN_N);
  __syncthreads();
  float s = b[t];
  for (int i = 0; i < 64; ++i) s += mean[i] * W[i * 64 + t];
  ge[t] = s;
}

__global__ void make_pb(const float* __restrict__ ge, const float* __restrict__ ae,
                        const float* __restrict__ Wp1, const float* __restrict__ bp1,
                        float* __restrict__ pb) {
  int idx = blockIdx.x * blockDim.x + threadIdx.x;
  if (idx >= A_N * H_N) return;
  int a = idx >> 10, n = idx & 1023;
  float s = bp1[n];
  for (int gi = 0; gi < G_N; ++gi)   s += ge[gi] * Wp1[(OBS_N + gi) * H_N + n];
  for (int e = 0; e < AEMB_N; ++e)   s += ae[a * AEMB_N + e] * Wp1[(OBS_N + G_N + e) * H_N + n];
  pb[idx] = s;
}

__global__ void make_vb(const float* __restrict__ ge, const float* __restrict__ Wv1,
                        const float* __restrict__ bv1, float* __restrict__ vb) {
  int n = blockIdx.x * blockDim.x + threadIdx.x;
  if (n >= H_N) return;
  float s = bv1[n];
  for (int gi = 0; gi < G_N; ++gi) s += ge[gi] * Wv1[(A_N * OBS_N + gi) * H_N + n];
  vb[n] = s;
}

// ---------------------------------------------------------------- WMMA GEMM
// C[M][Nn] = act( A_bf16 @ Wt_bf16^T + bias )
//   A elem(m,k) = Abase[ m*rowStride + (k>>kShift)*chunkStride + (k & ((1<<kShift)-1)) ]
//   Wt is N-major: Wt[n][k].  bias index = (m>>biasShift)*Nn + n.
// Async double-buffered LDS pipeline: prefetch tile kt+1 while computing kt.
#define BM 128
#define BN 128
#define BK 32

__global__ __launch_bounds__(256) void gemm_bf16_wmma(
    const unsigned short* __restrict__ Abase, int rowStride, int kShift, long long chunkStride,
    const unsigned short* __restrict__ Wt,
    const float* __restrict__ bias, int biasShift,
    float* __restrict__ outF, unsigned short* __restrict__ outH,
    int M, int Nn, int K, int relu) {
  __shared__ __align__(16) unsigned short As[2][BM * BK];
  __shared__ __align__(16) unsigned short Bs[2][BN * BK];

  const int m0 = blockIdx.y * BM;
  const int n0 = blockIdx.x * BN;
  const int tid = threadIdx.x;
  const int lane = tid & 31, wave = tid >> 5;
  const int wm = wave & 3;       // 4 wave-rows of 32
  const int wn = wave >> 2;      // 2 wave-cols of 64
  const int lr = lane & 15;
  const int koff = (lane >> 4) * 8;
  const int kMask = (1 << kShift) - 1;

  // staging geometry: 512 x 16B per tile, 2 per thread
  const int r0 = tid >> 1;                 // rows 0..127 (even idx)
  const int c0 = (tid & 1) * 16;           // 0 or 16  -> idx pattern: (tid, tid+256)
  // thread t handles (row=t>>2, col=(t&3)*8) and (row=(t>>2)+64, col=(t&3)*8)
  // simpler: idx = tid + it*256 ; r = idx>>2 ; c = (idx&3)*8
  (void)r0; (void)c0;

  // per-thread invariant pieces of the two staged elements
  long long aRowOff[2];
  const unsigned short* bSrcRow[2];
  int cCol[2];
#pragma unroll
  for (int it = 0; it < 2; ++it) {
    int idx = tid + it * 256;
    int r = idx >> 2;
    cCol[it] = (idx & 3) * 8;
    aRowOff[it] = (long long)(m0 + r) * rowStride;
    int gn = n0 + r; if (gn > Nn - 1) gn = Nn - 1;   // clamp (dup rows never stored)
    bSrcRow[it] = Wt + (long long)gn * K;
  }

  union FragU { v16bf v; v8bf h[2]; };
  const v8f vzero = {0.f, 0.f, 0.f, 0.f, 0.f, 0.f, 0.f, 0.f};
  v8f acc[2][4];
#pragma unroll
  for (int i = 0; i < 2; ++i)
#pragma unroll
    for (int j = 0; j < 4; ++j) acc[i][j] = vzero;

  const int nK = K / BK;

  auto stage = [&](int kt, int buf) {
    const int k0 = kt * BK;
#pragma unroll
    for (int it = 0; it < 2; ++it) {
      int idx = tid + it * 256;
      int r = idx >> 2, c = cCol[it];
      int gk = k0 + c;
      const unsigned short* asrc =
          Abase + aRowOff[it] + (long long)(gk >> kShift) * chunkStride + (gk & kMask);
      async_ld_b128(&As[buf][r * BK + c], asrc);
      async_ld_b128(&Bs[buf][r * BK + c], bSrcRow[it] + k0 + c);
    }
  };

  stage(0, 0);
  wait_async0();
  __syncthreads();

  for (int kt = 0; kt < nK; ++kt) {
    const int cur = kt & 1;
    if (kt + 1 < nK) stage(kt + 1, cur ^ 1);

    FragU fa[2], fb[4];
#pragma unroll
    for (int i = 0; i < 2; ++i) {
      int m = wm * 32 + i * 16 + lr;
      fa[i].h[0] = *(const v8bf*)&As[cur][m * BK + koff];
      fa[i].h[1] = *(const v8bf*)&As[cur][m * BK + koff + 16];
    }
#pragma unroll
    for (int j = 0; j < 4; ++j) {
      int n = wn * 64 + j * 16 + lr;
      fb[j].h[0] = *(const v8bf*)&Bs[cur][n * BK + koff];
      fb[j].h[1] = *(const v8bf*)&Bs[cur][n * BK + koff + 16];
    }
#pragma unroll
    for (int i = 0; i < 2; ++i)
#pragma unroll
      for (int j = 0; j < 4; ++j)
        acc[i][j] = __builtin_amdgcn_wmma_f32_16x16x32_bf16(
            false, fa[i].v, false, fb[j].v, (short)0, acc[i][j], false, false);

    wait_async0();      // prefetched tile landed
    __syncthreads();    // everyone done reading cur / writing next
  }

  // epilogue: C/D layout — VGPR r, lanes 0-15 => M=r, lanes 16-31 => M=r+8; N = lane%16
  const int mhalf = (lane >> 4) * 8;
#pragma unroll
  for (int i = 0; i < 2; ++i) {
#pragma unroll
    for (int j = 0; j < 4; ++j) {
      int n = n0 + wn * 64 + j * 16 + lr;
      if (n >= Nn) continue;
#pragma unroll
      for (int r = 0; r < 8; ++r) {
        int m = m0 + wm * 32 + i * 16 + mhalf + r;
        float v = acc[i][j][r] + bias[(long long)(m >> biasShift) * Nn + n];
        if (relu) v = fmaxf(v, 0.0f);
        long long o = (long long)m * Nn + n;
        if (outF) outF[o] = v;
        else      outH[o] = f2bf(v);
      }
    }
  }
}

// value head: out[b] = hv2[b] . Wv3 + bv3   (one wave per row)
__global__ __launch_bounds__(256) void value_out(const unsigned short* __restrict__ hv2,
                                                 const float* __restrict__ Wv3,
                                                 const float* __restrict__ bv3,
                                                 float* __restrict__ out, int M, int K) {
  int row = blockIdx.x * 8 + (threadIdx.x >> 5);
  int lane = threadIdx.x & 31;
  if (row >= M) return;
  float s = 0.f;
  for (int k = lane; k < K; k += 32) s += bf2f(hv2[(long long)row * K + k]) * Wv3[k];
#pragma unroll
  for (int off = 16; off > 0; off >>= 1) s += __shfl_xor(s, off, 32);
  if (lane == 0) out[row] = s + bv3[0];
}

// ---------------------------------------------------------------- launcher
extern "C" void kernel_launch(void* const* d_in, const int* in_sizes, int n_in,
                              void* d_out, int out_size, void* d_ws, size_t ws_size,
                              hipStream_t stream) {
  const float* obs   = (const float*)d_in[0];
  const int*   nt    = (const int*)d_in[1];
  const int*   na    = (const int*)d_in[2];
  const int*   edges = (const int*)d_in[3];
  const float* te    = (const float*)d_in[4];
  const float* gae   = (const float*)d_in[5];
  const float* W_l0  = (const float*)d_in[6];
  const float* b_l0  = (const float*)d_in[7];
  const float* g_l0  = (const float*)d_in[8];
  const float* be_l0 = (const float*)d_in[9];
  const float* W_l1  = (const float*)d_in[10];
  const float* b_l1  = (const float*)d_in[11];
  const float* g_l1  = (const float*)d_in[12];
  const float* be_l1 = (const float*)d_in[13];
  const float* W_gp  = (const float*)d_in[14];
  const float* b_gp  = (const float*)d_in[15];
  const float* aemb  = (const float*)d_in[16];
  const float* Wp1   = (const float*)d_in[17];
  const float* bp1   = (const float*)d_in[18];
  const float* Wp2   = (const float*)d_in[19];
  const float* bp2   = (const float*)d_in[20];
  const float* Wp3   = (const float*)d_in[21];
  const float* bp3   = (const float*)d_in[22];
  const float* Wv1   = (const float*)d_in[23];
  const float* bv1   = (const float*)d_in[24];
  const float* Wv2   = (const float*)d_in[25];
  const float* bv2   = (const float*)d_in[26];
  const float* Wv3   = (const float*)d_in[27];
  const float* bv3   = (const float*)d_in[28];

  // ---- workspace carving (256B aligned)
  char* ws = (char*)d_ws;
  size_t off = 0;
  auto carve = [&](size_t bytes) -> char* {
    char* p = ws + off;
    off = (off + bytes + 255) & ~(size_t)255;
    return p;
  };
  float* h0     = (float*)carve((size_t)NN_N * 64 * 4);
  float* agg    = (float*)carve((size_t)NN_N * 64 * 4);
  float* deg    = (float*)carve((size_t)NN_N * 4);
  float* h1     = (float*)carve((size_t)NN_N * 64 * 4);
  float* colsum = (float*)carve(64 * 4);
  float* ge     = (float*)carve(64 * 4);
  float* pb     = (float*)carve((size_t)A_N * H_N * 4);
  float* vb     = (float*)carve((size_t)H_N * 4);
  unsigned short* WtP1  = (unsigned short*)carve((size_t)H_N * OBS_N * 2);          // [1024][512]
  unsigned short* WtP2  = (unsigned short*)carve((size_t)H_N * H_N * 2);            // [1024][1024]
  unsigned short* WtP3  = (unsigned short*)carve((size_t)ACT_N * H_N * 2);          // [64][1024]
  unsigned short* WtV1  = (unsigned short*)carve((size_t)H_N * A_N * OBS_N * 2);    // [1024][4096]
  unsigned short* WtV2  = (unsigned short*)carve((size_t)H_N * H_N * 2);            // [1024][1024]
  unsigned short* obs16 = (unsigned short*)carve((size_t)A_N * B_N * OBS_N * 2);    // 32 MiB
  unsigned short* hpA   = (unsigned short*)carve((size_t)A_N * B_N * H_N * 2);      // 64 MiB
  unsigned short* hpB   = (unsigned short*)carve((size_t)A_N * B_N * H_N * 2);      // 64 MiB
  unsigned short* hv1   = hpA;  // value net reuses policy buffers (stream-serialized)
  unsigned short* hv2   = hpB;

  float* logits = (float*)d_out;                                  // 32768 x 64
  float* value  = (float*)d_out + (size_t)A_N * B_N * ACT_N;      // 4096

  // ---- 1) GNN
  zero_f32<<<(NN_N * 64 + 255) / 256, 256, 0, stream>>>(agg, NN_N * 64);
  zero_f32<<<(NN_N + 255) / 256, 256, 0, stream>>>(deg, NN_N);
  zero_f32<<<1, 64, 0, stream>>>(colsum, 64);

  gnn_l0<<<NN_N, 64, 0, stream>>>(nt, na, te, gae, W_l0, b_l0, g_l0, be_l0, h0);
  edge_scatter<<<(E_N * 64 + 255) / 256, 256, 0, stream>>>(edges, h0, agg, deg);
  gnn_l1<<<NN_N, 64, 0, stream>>>(h0, agg, deg, W_l1, b_l1, g_l1, be_l1, h1, colsum);
  graph_proj<<<1, 64, 0, stream>>>(colsum, W_gp, b_gp, ge);
  make_pb<<<(A_N * H_N + 255) / 256, 256, 0, stream>>>(ge, aemb, Wp1, bp1, pb);
  make_vb<<<(H_N + 255) / 256, 256, 0, stream>>>(ge, Wv1, bv1, vb);

  // ---- 2) weight prep: bf16, N-major
  {
    int n;
    n = OBS_N * H_N;         transpose_to_bf16<<<(n + 255) / 256, 256, 0, stream>>>(Wp1, WtP1, OBS_N, H_N);
    n = H_N * H_N;           transpose_to_bf16<<<(n + 255) / 256, 256, 0, stream>>>(Wp2, WtP2, H_N, H_N);
    n = H_N * ACT_N;         transpose_to_bf16<<<(n + 255) / 256, 256, 0, stream>>>(Wp3, WtP3, H_N, ACT_N);
    n = A_N * OBS_N * H_N;   transpose_to_bf16<<<(n + 255) / 256, 256, 0, stream>>>(Wv1, WtV1, A_N * OBS_N, H_N);
    n = H_N * H_N;           transpose_to_bf16<<<(n + 255) / 256, 256, 0, stream>>>(Wv2, WtV2, H_N, H_N);
  }
  {
    int n = A_N * B_N * OBS_N;
    f32_to_bf16_kernel<<<(n + 255) / 256, 256, 0, stream>>>(obs, obs16, n);
  }

  const int MP = A_N * B_N;   // 32768  (2^15)
  const int MV = B_N;         // 4096   (2^12)

  // ---- 3) policy MLP (bf16 WMMA)
  // hp1 = relu(obs @ Wp1[:512] + pb[a])        biasShift: 4096 -> 12
  gemm_bf16_wmma<<<dim3(H_N / BN, MP / BM), 256, 0, stream>>>(
      obs16, OBS_N, 9, 0, WtP1, pb, 12, nullptr, hpA, MP, H_N, OBS_N, 1);
  // hp2 = relu(hp1 @ Wp2 + bp2)                biasShift: 32768 -> 15
  gemm_bf16_wmma<<<dim3(H_N / BN, MP / BM), 256, 0, stream>>>(
      hpA, H_N, 10, 0, WtP2, bp2, 15, nullptr, hpB, MP, H_N, H_N, 1);
  // logits = hp2 @ Wp3 + bp3   (f32 straight to d_out)
  gemm_bf16_wmma<<<dim3((ACT_N + BN - 1) / BN, MP / BM), 256, 0, stream>>>(
      hpB, H_N, 10, 0, WtP3, bp3, 15, logits, nullptr, MP, ACT_N, H_N, 0);

  // ---- 4) value MLP (joint = obs transposed via chunked A addressing)
  // elem(b,k) = obs16[(k>>9)*B*512 + b*512 + (k&511)]
  gemm_bf16_wmma<<<dim3(H_N / BN, MV / BM), 256, 0, stream>>>(
      obs16, OBS_N, 9, (long long)B_N * OBS_N, WtV1, vb, 12, nullptr, hv1, MV, H_N, A_N * OBS_N, 1);
  // hv2 = relu(hv1 @ Wv2 + bv2)                biasShift: 4096 -> 12
  gemm_bf16_wmma<<<dim3(H_N / BN, MV / BM), 256, 0, stream>>>(
      hv1, H_N, 10, 0, WtV2, bv2, 12, nullptr, hv2, MV, H_N, H_N, 1);
  // value = hv2 @ Wv3 + bv3
  value_out<<<(MV + 7) / 8, 256, 0, stream>>>(hv2, Wv3, bv3, value, MV, H_N);

  (void)in_sizes; (void)n_in; (void)out_size; (void)ws_size;
}
